// MoELayer_84954453115202
// MI455X (gfx1250) — compile-verified, matrix operand order checked by reading, not run on hardware
//
#include <hip/hip_runtime.h>
#include <cstdint>

// ---------------- problem constants ----------------
#define T_TOK 4096          // B*S = 2*2048 tokens
#define HDIM  1024
#define IDIM  2048
#define NEXP  8
#define MAXROWS 12800       // 4096 shared + 8192 routed + 8*63 pad, rounded to 64
#define SUPER (MAXROWS/64)  // 200 M-supertiles

typedef __attribute__((ext_vector_type(16))) __bf16 v16bf;
typedef __attribute__((ext_vector_type(4)))  __bf16 v4bf;
typedef __attribute__((ext_vector_type(8)))  float  v8f;

__device__ __forceinline__ float clamp500(float v) { return fminf(fmaxf(v, -500.f), 500.f); }

// Load one 16-lane-striped bf16 fragment chunk for A/B of wmma_f32_16x16x32_bf16.
// Caller passes p = rowptr + k0 + (hi?8:0); layout: elems [0..7] at p, [8..15] at p+16.
__device__ __forceinline__ v16bf load_frag16(const __bf16* p) {
  union { v16bf v; uint4 q[2]; } u;
  u.q[0] = *(const uint4*)(p);
  u.q[1] = *(const uint4*)(p + 16);
  return u.v;
}

// ---------------- utility kernels ----------------
__global__ void k_zero(unsigned int* __restrict__ p, int n) {
  for (int i = blockIdx.x * blockDim.x + threadIdx.x; i < n; i += gridDim.x * blockDim.x)
    p[i] = 0u;
}

__global__ void k_cvt(const float* __restrict__ s, __bf16* __restrict__ d, int n) {
  for (int i = blockIdx.x * blockDim.x + threadIdx.x; i < n; i += gridDim.x * blockDim.x)
    d[i] = (__bf16)s[i];
}

// ---------------- router: LN + logits + softmax + top2 + aux stats ----------------
__global__ __launch_bounds__(256) void k_router(
    const float* __restrict__ x, const float* __restrict__ lnw, const float* __restrict__ lnb,
    const float* __restrict__ rw, const float* __restrict__ sgate,
    __bf16* __restrict__ hf_b, int* __restrict__ assign_e, float* __restrict__ assign_w,
    int* __restrict__ counts, int* __restrict__ tok, float* __restrict__ wtl,
    float* __restrict__ aux)
{
  const int lane = threadIdx.x & 31;
  const int t = blockIdx.x * 8 + (threadIdx.x >> 5);
  if (t >= T_TOK) return;
  const float* row = x + (size_t)t * HDIM;

  float v[32];
  float s = 0.f;
#pragma unroll
  for (int c = 0; c < 8; ++c) {
    float4 q = ((const float4*)row)[c * 32 + lane];
    float a0 = clamp500(q.x), a1 = clamp500(q.y), a2 = clamp500(q.z), a3 = clamp500(q.w);
    v[c*4+0] = a0; v[c*4+1] = a1; v[c*4+2] = a2; v[c*4+3] = a3;
    s += a0 + a1 + a2 + a3;
  }
#pragma unroll
  for (int o = 16; o > 0; o >>= 1) s += __shfl_xor(s, o, 32);
  const float mu = s * (1.f / HDIM);

  float s2 = 0.f;
#pragma unroll
  for (int j = 0; j < 32; ++j) { float d = v[j] - mu; s2 += d * d; }
#pragma unroll
  for (int o = 16; o > 0; o >>= 1) s2 += __shfl_xor(s2, o, 32);
  const float rs = rsqrtf(s2 * (1.f / HDIM) + 1e-5f);

  float lg[8];
#pragma unroll
  for (int e = 0; e < 8; ++e) lg[e] = 0.f;

  __bf16* hrow = hf_b + (size_t)t * HDIM;
#pragma unroll
  for (int c = 0; c < 8; ++c) {
    const int j0 = c * 128 + lane * 4;
    float4 w4 = ((const float4*)lnw)[j0 >> 2];
    float4 b4 = ((const float4*)lnb)[j0 >> 2];
    float h0 = fminf(fmaxf((v[c*4+0]-mu)*rs*w4.x + b4.x, -50.f), 50.f);
    float h1 = fminf(fmaxf((v[c*4+1]-mu)*rs*w4.y + b4.y, -50.f), 50.f);
    float h2 = fminf(fmaxf((v[c*4+2]-mu)*rs*w4.z + b4.z, -50.f), 50.f);
    float h3 = fminf(fmaxf((v[c*4+3]-mu)*rs*w4.w + b4.w, -50.f), 50.f);
#pragma unroll
    for (int e = 0; e < 8; ++e) {
      float4 r4 = ((const float4*)(rw + e * HDIM))[j0 >> 2];
      lg[e] += h0 * r4.x + h1 * r4.y + h2 * r4.z + h3 * r4.w;
    }
    v4bf pk;
    pk.x = (__bf16)v[c*4+0]; pk.y = (__bf16)v[c*4+1];
    pk.z = (__bf16)v[c*4+2]; pk.w = (__bf16)v[c*4+3];
    *(v4bf*)(hrow + j0) = pk;
  }

#pragma unroll
  for (int e = 0; e < 8; ++e) {
    float le = lg[e];
#pragma unroll
    for (int o = 16; o > 0; o >>= 1) le += __shfl_xor(le, o, 32);
    lg[e] = fminf(fmaxf(le, -10.f), 10.f);
  }

  if (lane == 0) {
    float mx = lg[0];
#pragma unroll
    for (int e = 1; e < 8; ++e) mx = fmaxf(mx, lg[e]);
    float pe[8], ssum = 0.f;
#pragma unroll
    for (int e = 0; e < 8; ++e) { pe[e] = expf(lg[e] - mx); ssum += pe[e]; }
    const float lse = mx + logf(ssum);
    float pr[8];
#pragma unroll
    for (int e = 0; e < 8; ++e) pr[e] = fminf(fmaxf(pe[e] / ssum, 1e-4f), 1.f);

    int e0 = 0;
#pragma unroll
    for (int e = 1; e < 8; ++e) if (pr[e] > pr[e0]) e0 = e;
    int e1 = (e0 == 0) ? 1 : 0;
#pragma unroll
    for (int e = 0; e < 8; ++e) if (e != e0 && pr[e] > pr[e1]) e1 = e;

    const float den = fmaxf(pr[e0] + pr[e1], 1e-4f);
    assign_e[2*t]   = e0; assign_e[2*t+1] = e1;
    assign_w[2*t]   = pr[e0] / den;
    assign_w[2*t+1] = pr[e1] / den;
    atomicAdd(&counts[e0], 1);
    atomicAdd(&counts[e1], 1);

    float ent = 0.f;
#pragma unroll
    for (int e = 0; e < 8; ++e) {
      atomicAdd(&aux[e], pr[e]);
      float ps = fminf(fmaxf(pr[e], 1e-4f), 1.f - 1e-4f);
      ent -= ps * logf(ps);
    }
    atomicAdd(&aux[8], lse * lse);
    atomicAdd(&aux[9], ent);

    // shared-expert fixed region [0, T): one slot per token, weight = sigmoid(gate)
    tok[t] = t;
    wtl[t] = 1.f / (1.f + expf(-sgate[0]));
  }
}

// ---------------- prefix offsets (padded to 64) ----------------
__global__ void k_offsets(const int* __restrict__ counts, int* __restrict__ offs) {
  if (threadIdx.x == 0 && blockIdx.x == 0) {
    offs[0] = 0;
    offs[1] = T_TOK;
    int a = T_TOK;
    for (int e = 0; e < 8; ++e) {
      a += ((counts[e] + 63) >> 6) << 6;
      offs[e + 2] = a;
    }
  }
}

__global__ void k_scatter(const int* __restrict__ assign_e, const float* __restrict__ assign_w,
                          const int* __restrict__ offs, int* __restrict__ cursor,
                          int* __restrict__ tok, float* __restrict__ wtl)
{
  int t = blockIdx.x * blockDim.x + threadIdx.x;
  if (t >= T_TOK) return;
#pragma unroll
  for (int k = 0; k < 2; ++k) {
    int e = assign_e[2*t + k];
    int p = atomicAdd(&cursor[e], 1);
    int b = offs[e + 1];
    tok[b + p] = t;
    wtl[b + p] = assign_w[2*t + k];
  }
}

// ---------------- GEMM1: gathered [64 x H] @ {wg,wu}^T -> silu*up -> prod bf16 ----------------
__global__ __launch_bounds__(256) void k_gemm1(
    const __bf16* __restrict__ hf_b, const __bf16* __restrict__ wg_b,
    const __bf16* __restrict__ wu_b, const int* __restrict__ tok,
    const int* __restrict__ offs, const int* __restrict__ counts,
    __bf16* __restrict__ prod)
{
  const int row0 = blockIdx.x * 64;
  if (row0 >= offs[9]) return;
  int ve = 0;
#pragma unroll
  for (int i = 1; i < 9; ++i) if (row0 >= offs[i]) ve = i;
  const int slot = (ve == 0) ? 8 : (ve - 1);
  const int base = offs[ve];
  const int cnt  = (ve == 0) ? T_TOK : counts[ve - 1];

  const __bf16* wgp = wg_b + (size_t)slot * IDIM * HDIM;
  const __bf16* wup = wu_b + (size_t)slot * IDIM * HDIM;

  const int lane = threadIdx.x & 31;
  const int wv   = threadIdx.x >> 5;
  const int r    = lane & 15;
  const int hi   = lane >> 4;
  const int icol = blockIdx.y * 128 + wv * 16 + r;

  const __bf16* wgrow = wgp + (size_t)icol * HDIM;
  const __bf16* wurow = wup + (size_t)icol * HDIM;

  const __bf16* arow[4];
#pragma unroll
  for (int m = 0; m < 4; ++m) {
    int idx = row0 + m * 16 + r;
    int tk = ((idx - base) < cnt) ? tok[idx] : 0;   // guard: padding slots hold poison
    arow[m] = hf_b + (size_t)tk * HDIM;
  }

  v8f cg[4], cu[4];
#pragma unroll
  for (int m = 0; m < 4; ++m)
#pragma unroll
    for (int j = 0; j < 8; ++j) { cg[m][j] = 0.f; cu[m][j] = 0.f; }

  for (int kk = 0; kk < HDIM; kk += 32) {
    const int kb = kk + hi * 8;
    v16bf bg = load_frag16(wgrow + kb);
    v16bf bu = load_frag16(wurow + kb);
#pragma unroll
    for (int m = 0; m < 4; ++m) {
      v16bf a = load_frag16(arow[m] + kb);
      cg[m] = __builtin_amdgcn_wmma_f32_16x16x32_bf16(false, a, false, bg, (short)0, cg[m], false, false);
      cu[m] = __builtin_amdgcn_wmma_f32_16x16x32_bf16(false, a, false, bu, (short)0, cu[m], false, false);
    }
  }

#pragma unroll
  for (int m = 0; m < 4; ++m) {
#pragma unroll
    for (int rr = 0; rr < 8; ++rr) {
      float g = clamp500(cg[m][rr]);
      g = g / (1.f + expf(-g));          // silu(clamp(gate))
      float u = clamp500(cu[m][rr]);
      float pr = clamp500(g * u);
      int mrow = row0 + m * 16 + rr + hi * 8;
      prod[(size_t)mrow * IDIM + icol] = (__bf16)pr;
    }
  }
}

// ---------------- GEMM2: prod @ wd^T, clamp, weight, atomic accumulate ----------------
__global__ __launch_bounds__(256) void k_gemm2(
    const __bf16* __restrict__ prod, const __bf16* __restrict__ wd_b,
    const int* __restrict__ tok, const float* __restrict__ wtl,
    const int* __restrict__ offs, const int* __restrict__ counts,
    float* __restrict__ acc)
{
  const int row0 = blockIdx.x * 64;
  if (row0 >= offs[9]) return;
  int ve = 0;
#pragma unroll
  for (int i = 1; i < 9; ++i) if (row0 >= offs[i]) ve = i;
  const int slot = (ve == 0) ? 8 : (ve - 1);
  const int base = offs[ve];
  const int cnt  = (ve == 0) ? T_TOK : counts[ve - 1];

  const __bf16* wdp = wd_b + (size_t)slot * HDIM * IDIM;

  const int lane = threadIdx.x & 31;
  const int wv   = threadIdx.x >> 5;
  const int r    = lane & 15;
  const int hi   = lane >> 4;
  const int hcol = blockIdx.y * 128 + wv * 16 + r;

  const __bf16* wdrow = wdp + (size_t)hcol * IDIM;
  const __bf16* arow[4];
#pragma unroll
  for (int m = 0; m < 4; ++m)
    arow[m] = prod + (size_t)(row0 + m * 16 + r) * IDIM;

  v8f c[4];
#pragma unroll
  for (int m = 0; m < 4; ++m)
#pragma unroll
    for (int j = 0; j < 8; ++j) c[m][j] = 0.f;

  for (int kk = 0; kk < IDIM; kk += 32) {
    const int kb = kk + hi * 8;
    v16bf b = load_frag16(wdrow + kb);
#pragma unroll
    for (int m = 0; m < 4; ++m) {
      v16bf a = load_frag16(arow[m] + kb);
      c[m] = __builtin_amdgcn_wmma_f32_16x16x32_bf16(false, a, false, b, (short)0, c[m], false, false);
    }
  }

#pragma unroll
  for (int m = 0; m < 4; ++m) {
#pragma unroll
    for (int rr = 0; rr < 8; ++rr) {
      int idx = row0 + m * 16 + rr + hi * 8;
      if ((idx - base) < cnt) {
        int t = tok[idx];
        float w = wtl[idx];
        atomicAdd(&acc[(size_t)t * HDIM + hcol], w * clamp500(c[m][rr]));
      }
    }
  }
}

// ---------------- finalize: clamp output + aux scalar ----------------
__global__ void k_final(const float* __restrict__ acc, float* __restrict__ out,
                        const int* __restrict__ counts, const float* __restrict__ aux)
{
  const int n = T_TOK * HDIM;
  for (int i = blockIdx.x * blockDim.x + threadIdx.x; i < n; i += gridDim.x * blockDim.x)
    out[i] = clamp500(acc[i]);

  if (blockIdx.x == 0 && threadIdx.x == 0) {
    float lb = 0.f, usage = 0.f;
#pragma unroll
    for (int e = 0; e < 8; ++e) {
      float tpe = counts[e] / (float)(T_TOK * 2);
      float avg = aux[e] / (float)T_TOK;
      lb += tpe * avg;
      usage += (tpe > 0.01f) ? 1.f : 0.f;
    }
    lb *= 8.f;
    float z    = (aux[8] / (float)T_TOK) * 0.001f;
    float ent  = aux[9] / (float)T_TOK;
    float entl = fmaxf(logf(8.f) - ent, 0.f) * 0.01f;
    float util = (1.f - usage / 8.f) * 0.1f;
    float a = lb + z + entl + util;
    out[n] = fminf(fmaxf(a, 0.f), 10.f);
  }
}

// ---------------- host launcher ----------------
extern "C" void kernel_launch(void* const* d_in, const int* in_sizes, int n_in,
                              void* d_out, int out_size, void* d_ws, size_t ws_size,
                              hipStream_t stream) {
  const float* x     = (const float*)d_in[0];
  const float* ln_w  = (const float*)d_in[1];
  const float* ln_b  = (const float*)d_in[2];
  const float* rw    = (const float*)d_in[3];
  const float* wg    = (const float*)d_in[4];
  const float* wu    = (const float*)d_in[5];
  const float* wd    = (const float*)d_in[6];
  const float* swg   = (const float*)d_in[7];
  const float* swu   = (const float*)d_in[8];
  const float* swd   = (const float*)d_in[9];
  const float* sgate = (const float*)d_in[10];
  float* out = (float*)d_out;

  char* ws = (char*)d_ws;
  size_t cur = 0;
  auto alloc = [&](size_t bytes) {
    size_t o = cur;
    cur = (cur + bytes + 255) & ~(size_t)255;
    return o;
  };

  const size_t perW = (size_t)IDIM * HDIM;     // elements in one expert's wg/wu slice
  size_t WGB = alloc(9 * perW * 2);
  size_t WUB = alloc(9 * perW * 2);
  size_t WDB = alloc(9 * perW * 2);
  size_t HFB = alloc((size_t)T_TOK * HDIM * 2);
  size_t PRD = alloc((size_t)MAXROWS * IDIM * 2);
  size_t ACC = alloc((size_t)T_TOK * HDIM * 4);
  size_t TOK = alloc((size_t)MAXROWS * 4);
  size_t WTL = alloc((size_t)MAXROWS * 4);
  size_t AE  = alloc((size_t)T_TOK * 2 * 4);
  size_t AW  = alloc((size_t)T_TOK * 2 * 4);
  size_t MET = alloc(64 * 4);                  // counts[8] | cursor[8] | offs[16] | aux[10]

  __bf16* wg_b = (__bf16*)(ws + WGB);
  __bf16* wu_b = (__bf16*)(ws + WUB);
  __bf16* wd_b = (__bf16*)(ws + WDB);
  __bf16* hf_b = (__bf16*)(ws + HFB);
  __bf16* prod = (__bf16*)(ws + PRD);
  float*  accp = (float*)(ws + ACC);
  int*    tokp = (int*)(ws + TOK);
  float*  wtlp = (float*)(ws + WTL);
  int*    ae   = (int*)(ws + AE);
  float*  aw   = (float*)(ws + AW);
  int*    counts = (int*)(ws + MET);
  int*    cursor = counts + 8;
  int*    offs   = cursor + 8;
  float*  aux    = (float*)(offs + 16);

  // zero accumulators + metadata
  k_zero<<<2048, 256, 0, stream>>>((unsigned int*)accp, T_TOK * HDIM);
  k_zero<<<1, 64, 0, stream>>>((unsigned int*)counts, 64);

  // convert weights f32 -> bf16 (shared expert in slot 8)
  k_cvt<<<4096, 256, 0, stream>>>(wg,  wg_b,                 8 * (int)perW);
  k_cvt<<<2048, 256, 0, stream>>>(swg, wg_b + 8 * perW,      (int)perW);
  k_cvt<<<4096, 256, 0, stream>>>(wu,  wu_b,                 8 * (int)perW);
  k_cvt<<<2048, 256, 0, stream>>>(swu, wu_b + 8 * perW,      (int)perW);
  k_cvt<<<4096, 256, 0, stream>>>(wd,  wd_b,                 8 * (int)perW);
  k_cvt<<<2048, 256, 0, stream>>>(swd, wd_b + 8 * perW,      (int)perW);

  // router + assignment build
  k_router<<<T_TOK / 8, 256, 0, stream>>>(x, ln_w, ln_b, rw, sgate, hf_b,
                                          ae, aw, counts, tokp, wtlp, aux);
  k_offsets<<<1, 32, 0, stream>>>(counts, offs);
  k_scatter<<<T_TOK / 256, 256, 0, stream>>>(ae, aw, offs, cursor, tokp, wtlp);

  // expert SwiGLU via bf16 WMMA
  dim3 g1(SUPER, IDIM / 128);
  k_gemm1<<<g1, 256, 0, stream>>>(hf_b, wg_b, wu_b, tokp, offs, counts, prod);
  dim3 g2(SUPER, HDIM / 128);
  k_gemm2<<<g2, 256, 0, stream>>>(prod, wd_b, tokp, wtlp, offs, counts, accp);

  // finalize output + aux
  k_final<<<4096, 256, 0, stream>>>(accp, out, counts, aux);

  (void)in_sizes; (void)n_in; (void)out_size; (void)ws_size;
}
